// MoERouter_55705725829208
// MI455X (gfx1250) — matrix operand order repaired and checked by reading, not compiled, not run on hardware
//
#include <hip/hip_runtime.h>
#include <hip/hip_fp16.h>

typedef _Float16 f16;
typedef __attribute__((ext_vector_type(16))) _Float16 v16h;
typedef __attribute__((ext_vector_type(8)))  float    v8f;
typedef __attribute__((ext_vector_type(4)))  unsigned int u32x4;
typedef __attribute__((ext_vector_type(8)))  int      i32x8;
typedef __attribute__((ext_vector_type(4)))  int      i32x4;

#define NTOK 16384
#define DIN  512
#define HID  512
#define OUTD 64
#define NEXP 8

#if __has_builtin(__builtin_amdgcn_tensor_load_to_lds) && __has_builtin(__builtin_amdgcn_s_wait_tensorcnt)
#define USE_TDM 1
#else
#define USE_TDM 0
#endif

union HV { uint4 u[2]; v16h h; };

#if USE_TDM
// Issue a TDM load of a 64-row x 32-col f16 tile (row stride 512 elements in
// global) into LDS at byte offset lds_off, padding LDS rows 64B->80B so the
// WMMA fragment reads below stay bank-conflict free.
// D# layout per CDNA5 ISA ch.8 (group0 128b, group1 256b, groups 2/3 unused).
__device__ __forceinline__ void tdm_load_tile(unsigned lds_off, const f16* gp) {
  unsigned long long ga = (unsigned long long)(uintptr_t)gp;
  u32x4 g0;
  g0[0] = 1u;                                   // count=1, user mode
  g0[1] = lds_off;                              // lds_addr
  g0[2] = (unsigned)(ga & 0xFFFFFFFFu);         // global_addr[31:0]
  g0[3] = (unsigned)((ga >> 32) & 0x01FFFFFFu)  // global_addr[56:32]
        | (2u << 30);                           // type = 2 (image)
  i32x8 g1;
  g1[0] = (1 << 16)      // data_size = 1 -> 2 bytes
        | (1 << 20)      // pad_enable
        | (3 << 22)      // pad_interval: 16 DWORDs (one 64B row)
        | (3 << 25);     // pad_amount:   4 DWORDs (16B) -> 80B LDS row stride
  g1[1] = (int)(512u << 16);    // tensor_dim0[15:0] = 512 (bits 79:48)
  g1[2] = (int)(16384u << 16);  // tensor_dim1[15:0] = NTOK (bits 111:80)
  g1[3] = (int)(32u << 16);     // tile_dim0 = 32 (bits 127:112)
  g1[4] = 64;                   // tile_dim1 = 64 (bits 143:128)
  g1[5] = 512;                  // tensor_dim0_stride = 512 (bits 207:160)
  g1[6] = 0;
  g1[7] = 0;
  i32x4 z4 = {};
#if __clang_major__ >= 23
  i32x8 z8 = {};
  __builtin_amdgcn_tensor_load_to_lds(g0, g1, z4, z4, z8, 0);
#else
  __builtin_amdgcn_tensor_load_to_lds(g0, g1, z4, z4, 0);
#endif
}
#endif

// ---------------------------------------------------------------------------
// GEMM (N x 512) = X(N x 512) @ W(512 x 512), fused bias + LayerNorm + GELU.
// WT is the f16 weight pre-transposed to [Hout=512][Kin=512].
// Block: 256 threads (8 wave32). Block tile: 64 rows x 512 cols.
// A tile staged into LDS by the Tensor Data Mover (double buffered, one
// barrier per K-chunk); B fragments read straight from L2-resident weights.
// ---------------------------------------------------------------------------
__global__ __launch_bounds__(256) void k_gemm_ln_gelu(
    const f16* __restrict__ X, const f16* __restrict__ WT,
    const float* __restrict__ bias, const float* __restrict__ lng,
    const float* __restrict__ lnb, f16* __restrict__ Y)
{
  __shared__ __align__(16) f16 As[2][64][40];   // 64 x 32 tiles, 80B row stride
  __shared__ float rsum[64], rsq[64];

  const int tid    = threadIdx.x;
  const int wave   = tid >> 5;
  const int lane   = tid & 31;
  const int half16 = lane >> 4;
  const int m      = lane & 15;
  const long rowbase = (long)blockIdx.x * 64;
  const int cb     = wave * 64;
  const int kbA    = half16 * 8;    // A: lanes 0-15 -> K 0-7 & 16-23; 16-31 -> K 8-15 & 24-31
  const int kbB    = half16 * 8;    // B: lanes 0-15 -> K 0-7 & 16-23; 16-31 -> K 8-15 & 24-31

  v8f acc[4][4] = {};

#if USE_TDM
  const unsigned lds0 = (unsigned)(uintptr_t)&As[0][0][0];
  const unsigned lds1 = (unsigned)(uintptr_t)&As[1][0][0];
  if (tid < 32) tdm_load_tile(lds0, X + rowbase * 512);
#else
  const int ar = tid >> 2;
  const int ac = (tid & 3) * 8;
#endif

  for (int kk = 0; kk < 512; kk += 32) {
    const int sel = (kk >> 5) & 1;
#if USE_TDM
    if (tid < 32) __builtin_amdgcn_s_wait_tensorcnt((short)0);
    __syncthreads();                 // tile ready + prev reads of other buf done
    if (tid < 32 && kk + 32 < 512)
      tdm_load_tile(sel ? lds0 : lds1, X + rowbase * 512 + kk + 32);
#else
    (void)sel;
    *(uint4*)&As[0][ar][ac] = *(const uint4*)(X + (rowbase + ar) * 512 + kk + ac);
    __syncthreads();
#endif

    // B fragments: two 16B loads per lane (k kbB..kbB+7, kbB+16..kbB+23) from transposed weights (L2-resident)
    v16h bf[4];
#pragma unroll
    for (int ct = 0; ct < 4; ++ct) {
      const f16* wp = WT + (long)(cb + ct * 16 + m) * 512 + kk + kbB;
      HV t; t.u[0] = *(const uint4*)wp; t.u[1] = *(const uint4*)(wp + 16);
      bf[ct] = t.h;
    }
#pragma unroll
    for (int rt = 0; rt < 4; ++rt) {
#if USE_TDM
      const f16* ap = &As[sel][rt * 16 + m][kbA];
#else
      const f16* ap = &As[0][rt * 16 + m][kbA];
#endif
      HV a; a.u[0] = *(const uint4*)ap; a.u[1] = *(const uint4*)(ap + 16);
#pragma unroll
      for (int ct = 0; ct < 4; ++ct)
        acc[rt][ct] = __builtin_amdgcn_wmma_f32_16x16x32_f16(
            false, a.h, false, bf[ct], (short)0, acc[rt][ct], false, false);
    }
#if !USE_TDM
    __syncthreads();
#endif
  }

  // ---------------- epilogue: bias + LayerNorm + GELU -----------------------
  if (tid < 64) { rsum[tid] = 0.f; rsq[tid] = 0.f; }
  __syncthreads();

#pragma unroll
  for (int rt = 0; rt < 4; ++rt) {
#pragma unroll
    for (int i = 0; i < 8; ++i) {
      float s = 0.f, q = 0.f;
#pragma unroll
      for (int ct = 0; ct < 4; ++ct) {
        float v = acc[rt][ct][i] + bias[cb + ct * 16 + m];
        acc[rt][ct][i] = v;
        s += v; q += v * v;
      }
      // reduce across the 16 lanes that share this row (within a half-wave)
      for (int off = 1; off < 16; off <<= 1) {
        s += __shfl_xor(s, off, 16);
        q += __shfl_xor(q, off, 16);
      }
      if (m == 0) {
        int r = rt * 16 + half16 * 8 + i;
        atomicAdd(&rsum[r], s);
        atomicAdd(&rsq[r],  q);
      }
    }
  }
  __syncthreads();

#pragma unroll
  for (int rt = 0; rt < 4; ++rt) {
#pragma unroll
    for (int i = 0; i < 8; ++i) {
      int r = rt * 16 + half16 * 8 + i;
      float mu   = rsum[r] * (1.f / 512.f);
      float var  = rsq[r]  * (1.f / 512.f) - mu * mu;
      float rstd = rsqrtf(var + 1e-5f);
#pragma unroll
      for (int ct = 0; ct < 4; ++ct) {
        int   c = cb + ct * 16 + m;
        float x = (acc[rt][ct][i] - mu) * rstd * lng[c] + lnb[c];
        float g = 0.5f * x * (1.f + erff(x * 0.70710678118654752f));
        Y[(rowbase + r) * 512 + c] = (f16)g;
      }
    }
  }
}

// ---------------------------------------------------------------------------
// Expert output GEMM: (N x 64) = X(N x 512) @ W3(512 x 64), fused bias and
// gate-weight scaling accumulated into combined[N][64].
// Block: 8 waves, each wave = one 16-row tile x 64 cols. Block = 128 rows.
// ---------------------------------------------------------------------------
__global__ __launch_bounds__(256) void k_gemm_out_acc(
    const f16* __restrict__ X, const f16* __restrict__ WT,
    const float* __restrict__ bias, const float* __restrict__ wdense,
    int e, float* __restrict__ comb)
{
  const int tid = threadIdx.x, wave = tid >> 5, lane = tid & 31;
  const int half16 = lane >> 4, m = lane & 15;
  const long rowbase = (long)blockIdx.x * 128 + wave * 16;
  const int kbA = half16 * 8, kbB = half16 * 8;

  v8f acc[4] = {};
  for (int kk = 0; kk < 512; kk += 32) {
    const f16* ap = X + (rowbase + m) * 512 + kk + kbA;
    HV a; a.u[0] = *(const uint4*)ap; a.u[1] = *(const uint4*)(ap + 16);
    if (kk + 32 < 512)
      __builtin_prefetch(X + (rowbase + m) * 512 + kk + 32 + kbA, 0, 0);
#pragma unroll
    for (int ct = 0; ct < 4; ++ct) {
      const f16* wp = WT + (long)(ct * 16 + m) * 512 + kk + kbB;
      HV b; b.u[0] = *(const uint4*)wp; b.u[1] = *(const uint4*)(wp + 16);
      acc[ct] = __builtin_amdgcn_wmma_f32_16x16x32_f16(
          false, a.h, false, b.h, (short)0, acc[ct], false, false);
    }
  }
#pragma unroll
  for (int i = 0; i < 8; ++i) {
    long r = rowbase + half16 * 8 + i;
    float w = wdense[r * 8 + e];
#pragma unroll
    for (int ct = 0; ct < 4; ++ct) {
      int c = ct * 16 + m;
      comb[r * 64 + c] += w * (acc[ct][i] + bias[c]);
    }
  }
}

// ---------------------------------------------------------------------------
// Gate tail: logits (512x8), softmax, top-2, normalized weights, dense routing
// weights, expert usage.
// ---------------------------------------------------------------------------
__global__ __launch_bounds__(256) void k_gate(
    const f16* __restrict__ GH, const float* __restrict__ W2,
    const float* __restrict__ B2, float* __restrict__ gw_out,
    int* __restrict__ idx_out, float* __restrict__ tkw_out,
    float* __restrict__ usage_out, float* __restrict__ wdense)
{
  __shared__ float us[8];
  int tid = threadIdx.x;
  if (tid < 8) us[tid] = 0.f;
  __syncthreads();

  long n = (long)blockIdx.x * 256 + tid;
  float a[8];
#pragma unroll
  for (int j = 0; j < 8; ++j) a[j] = B2[j];
  for (int k = 0; k < 512; ++k) {
    float gv = (float)GH[n * 512 + k];
    const float4* wp = (const float4*)(W2 + k * 8);
    float4 w0 = wp[0], w1 = wp[1];
    a[0] += gv * w0.x; a[1] += gv * w0.y; a[2] += gv * w0.z; a[3] += gv * w0.w;
    a[4] += gv * w1.x; a[5] += gv * w1.y; a[6] += gv * w1.z; a[7] += gv * w1.w;
  }
  float mx = a[0];
#pragma unroll
  for (int j = 1; j < 8; ++j) mx = fmaxf(mx, a[j]);
  float s = 0.f;
#pragma unroll
  for (int j = 0; j < 8; ++j) { a[j] = expf(a[j] - mx); s += a[j]; }
  float inv = 1.f / s;
#pragma unroll
  for (int j = 0; j < 8; ++j) {
    a[j] *= inv;
    gw_out[n * 8 + j] = a[j];
    atomicAdd(&us[j], a[j]);
  }
  int i0 = 0; float w0 = a[0];
#pragma unroll
  for (int j = 1; j < 8; ++j) if (a[j] > w0) { w0 = a[j]; i0 = j; }
  int i1 = (i0 == 0) ? 1 : 0; float w1 = a[i1];
#pragma unroll
  for (int j = 0; j < 8; ++j)
    if (j != i0 && a[j] > w1) { w1 = a[j]; i1 = j; }
  float norm = 1.f / (w0 + w1);
  idx_out[n * 2]     = i0;
  idx_out[n * 2 + 1] = i1;
  tkw_out[n * 2]     = w0 * norm;
  tkw_out[n * 2 + 1] = w1 * norm;
#pragma unroll
  for (int j = 0; j < 8; ++j)
    wdense[n * 8 + j] = (j == i0) ? w0 * norm : ((j == i1) ? w1 * norm : 0.f);
  __syncthreads();
  if (tid < 8) atomicAdd(&usage_out[tid], us[tid] * (1.f / (float)NTOK));
}

// ---------------------------------------------------------------------------
// Final projection: out(N x 64) = combined(N x 64) @ p_w(64 x 64) + p_b.
// Tiny (0.1% of FLOPs) -> scalar f32.
// ---------------------------------------------------------------------------
__global__ __launch_bounds__(256) void k_proj(
    const float* __restrict__ comb, const float* __restrict__ PW,
    const float* __restrict__ PB, float* __restrict__ out)
{
  long id = (long)blockIdx.x * 256 + threadIdx.x;
  long n = id >> 6; int o = (int)(id & 63);
  float s = PB[o];
  const float* cr = comb + n * 64;
#pragma unroll 8
  for (int k = 0; k < 64; ++k) s += cr[k] * PW[k * 64 + o];
  out[n * 64 + o] = s;
}

// -------------------------- conversion / init -----------------------------
__global__ void k_cvt(const float* __restrict__ src, f16* __restrict__ dst, long n) {
  long i = (long)blockIdx.x * 256 + threadIdx.x;
  if (i < n) dst[i] = (f16)src[i];
}
// src [B][K][H] f32 row-major -> dst [B][H][K] f16 (per-batch transpose)
__global__ void k_cvtT(const float* __restrict__ src, f16* __restrict__ dst,
                       int K, int H, long total) {
  long i = (long)blockIdx.x * 256 + threadIdx.x;
  if (i >= total) return;
  long kh = (long)K * H;
  long b = i / kh;
  long r = i - b * kh;
  long k = r / H;
  long h = r - k * H;
  dst[b * kh + h * (long)K + k] = (f16)src[i];
}
__global__ void k_zero(float* p, long n) {
  long i = (long)blockIdx.x * 256 + threadIdx.x;
  if (i < n) p[i] = 0.f;
}

// ---------------------------------------------------------------------------
extern "C" void kernel_launch(void* const* d_in, const int* in_sizes, int n_in,
                              void* d_out, int out_size, void* d_ws, size_t ws_size,
                              hipStream_t stream) {
  (void)in_sizes; (void)n_in; (void)out_size; (void)ws_size;
  const float* x      = (const float*)d_in[0];
  const float* g_w1   = (const float*)d_in[1];
  const float* g_b1   = (const float*)d_in[2];
  const float* g_ln_g = (const float*)d_in[3];
  const float* g_ln_b = (const float*)d_in[4];
  const float* g_w2   = (const float*)d_in[5];
  const float* g_b2   = (const float*)d_in[6];
  const float* e_w1   = (const float*)d_in[7];
  const float* e_b1   = (const float*)d_in[8];
  const float* e_ln1g = (const float*)d_in[9];
  const float* e_ln1b = (const float*)d_in[10];
  const float* e_w2   = (const float*)d_in[11];
  const float* e_b2   = (const float*)d_in[12];
  const float* e_ln2g = (const float*)d_in[13];
  const float* e_ln2b = (const float*)d_in[14];
  const float* e_w3   = (const float*)d_in[15];
  const float* e_b3   = (const float*)d_in[16];
  const float* p_w    = (const float*)d_in[17];
  const float* p_b    = (const float*)d_in[18];

  char* ws = (char*)d_ws;
  size_t off = 0;
  f16*   xh     = (f16*)(ws + off);  off += (size_t)NTOK * DIN  * 2;
  f16*   act1   = (f16*)(ws + off);  off += (size_t)NTOK * HID  * 2;
  f16*   act2   = (f16*)(ws + off);  off += (size_t)NTOK * HID  * 2;
  float* comb   = (float*)(ws + off); off += (size_t)NTOK * OUTD * 4;
  float* wdense = (float*)(ws + off); off += (size_t)NTOK * NEXP * 4;
  f16*   g1T    = (f16*)(ws + off);  off += (size_t)HID * DIN * 2;
  f16*   e1T    = (f16*)(ws + off);  off += (size_t)NEXP * HID * DIN * 2;
  f16*   e2T    = (f16*)(ws + off);  off += (size_t)NEXP * HID * HID * 2;
  f16*   e3T    = (f16*)(ws + off);  off += (size_t)NEXP * OUTD * HID * 2;

  float* o_out   = (float*)d_out;
  float* o_gw    = o_out + (size_t)NTOK * OUTD;
  int*   o_idx   = (int*)(o_gw + (size_t)NTOK * NEXP);
  float* o_tkw   = (float*)(o_idx + (size_t)NTOK * 2);
  float* o_usage = o_tkw + (size_t)NTOK * 2;

  // init accumulators (d_out / d_ws are poisoned by harness)
  k_zero<<<(NTOK * OUTD + 255) / 256, 256, 0, stream>>>(comb, (long)NTOK * OUTD);
  k_zero<<<1, 256, 0, stream>>>(o_usage, 8);

  // f32 -> f16 conversions (weights transposed to [Hout][Kin])
  k_cvt<<<((long)NTOK * DIN + 255) / 256, 256, 0, stream>>>(x, xh, (long)NTOK * DIN);
  k_cvtT<<<((long)DIN * HID + 255) / 256, 256, 0, stream>>>(g_w1, g1T, DIN, HID, (long)DIN * HID);
  k_cvtT<<<((long)NEXP * DIN * HID + 255) / 256, 256, 0, stream>>>(e_w1, e1T, DIN, HID, (long)NEXP * DIN * HID);
  k_cvtT<<<((long)NEXP * HID * HID + 255) / 256, 256, 0, stream>>>(e_w2, e2T, HID, HID, (long)NEXP * HID * HID);
  k_cvtT<<<((long)NEXP * HID * OUTD + 255) / 256, 256, 0, stream>>>(e_w3, e3T, HID, OUTD, (long)NEXP * HID * OUTD);

  // gate
  k_gemm_ln_gelu<<<NTOK / 64, 256, 0, stream>>>(xh, g1T, g_b1, g_ln_g, g_ln_b, act1);
  k_gate<<<NTOK / 256, 256, 0, stream>>>(act1, g_w2, g_b2, o_gw, o_idx, o_tkw, o_usage, wdense);

  // experts (dense, all tokens)
  for (int e = 0; e < NEXP; ++e) {
    k_gemm_ln_gelu<<<NTOK / 64, 256, 0, stream>>>(
        xh, e1T + (size_t)e * HID * DIN, e_b1 + e * HID,
        e_ln1g + e * HID, e_ln1b + e * HID, act1);
    k_gemm_ln_gelu<<<NTOK / 64, 256, 0, stream>>>(
        act1, e2T + (size_t)e * HID * HID, e_b2 + e * HID,
        e_ln2g + e * HID, e_ln2b + e * HID, act2);
    k_gemm_out_acc<<<NTOK / 128, 256, 0, stream>>>(
        act2, e3T + (size_t)e * OUTD * HID, e_b3 + e * OUTD, wdense, e, comb);
  }

  // output projection
  k_proj<<<((long)NTOK * OUTD + 255) / 256, 256, 0, stream>>>(comb, p_w, p_b, o_out);
}
